// Multi_Head_Attention_55576876810425
// MI455X (gfx1250) — matrix operand order repaired and checked by reading, not compiled, not run on hardware
//
#include <hip/hip_runtime.h>
#include <stdint.h>

#define BATCH 4
#define SEQ   2048
#define EMB   1024
#define NH    16
#define HDIM  64

typedef __attribute__((ext_vector_type(8)))  float   v8f;
typedef __attribute__((ext_vector_type(16))) __bf16  v16bf;
typedef __attribute__((ext_vector_type(8)))  __bf16  v8bf;
typedef __attribute__((ext_vector_type(4)))  int     v4i;

// Native conversion -> compiler emits v_cvt_pk_bf16_f32 (RNE), not bit tricks.
__device__ __forceinline__ __bf16 tobf(float f) { return (__bf16)f; }

__device__ __forceinline__ v8f v8f_zero() {
  v8f z;
#pragma unroll
  for (int i = 0; i < 8; ++i) z[i] = 0.0f;
  return z;
}

__device__ __forceinline__ v8f wmma_bf16(v16bf a, v16bf b, v8f c) {
  // D = A(16x32 bf16) * B(32x16 bf16) + C(16x16 f32)
  return __builtin_amdgcn_wmma_f32_16x16x32_bf16(false, a, false, b, (short)0, c,
                                                 false, false);
}

// ---------------------------------------------------------------------------
// One-shot weight conversion f32 -> bf16 (weights are L2-resident and reused
// by every M-block; convert once instead of per-workgroup).
// ---------------------------------------------------------------------------
__global__ __launch_bounds__(256) void cvt_kernel(const float* __restrict__ src,
                                                  __bf16* __restrict__ dst, int n8)
{
  const int i = blockIdx.x * 256 + threadIdx.x;
  if (i >= n8) return;
  v8f x = *(const v8f*)(src + (size_t)i * 8);
  v8bf y;
#pragma unroll
  for (int j = 0; j < 8; ++j) y[j] = tobf(x[j]);
  *(v8bf*)(dst + (size_t)i * 8) = y;
}

// ---------------------------------------------------------------------------
// Projection GEMM:  outp = A(f32 [8192,1024]) @ W^T(bf16 [1024,1024]) + bias,
// stored bf16 in split-head layout [b,h,s,d] (or transposed [b,h,d,s] for V).
// Block: 256 thr = 8 waves; wave tile 32(M)x64(N); block tile 256x64.
// ---------------------------------------------------------------------------
__global__ __launch_bounds__(256) void proj_kernel(
    const float* __restrict__ A, const __bf16* __restrict__ W,
    const float* __restrict__ bias, __bf16* __restrict__ outp, int transposeOut)
{
  const int lane  = threadIdx.x & 31;
  const int wave  = threadIdx.x >> 5;
  const int mbase = blockIdx.x * 256 + wave * 32;
  const int nbase = blockIdx.y * 64;
  const int lm = lane & 15;   // row-in-tile for A/B fragments
  const int lh = lane >> 4;   // lane half

  v8f acc[2][4];
#pragma unroll
  for (int s = 0; s < 2; ++s)
#pragma unroll
    for (int t = 0; t < 4; ++t) acc[s][t] = v8f_zero();

  for (int kc = 0; kc < EMB; kc += 32) {
    // A fragments (convert f32 -> bf16 on the fly), layout:
    // e0..15: k = kc + lh*16 + e
    v16bf afr[2];
#pragma unroll
    for (int s = 0; s < 2; ++s) {
      const float* ar = A + (size_t)(mbase + s * 16 + lm) * EMB + kc + lh * 16;
      v8f lo = *(const v8f*)(ar);
      v8f hi = *(const v8f*)(ar + 8);
#pragma unroll
      for (int i = 0; i < 8; ++i) { afr[s][i] = tobf(lo[i]); afr[s][i + 8] = tobf(hi[i]); }
    }
    // B fragments: B(k,n) = W[n][kc+k]; lane holds n=lm, k = lh*16 + e
#pragma unroll
    for (int t = 0; t < 4; ++t) {
      const __bf16* wr = W + (size_t)(nbase + t * 16 + lm) * EMB + kc + lh * 16;
      v16bf bfr = *(const v16bf*)(wr);
      acc[0][t] = wmma_bf16(afr[0], bfr, acc[0][t]);
      acc[1][t] = wmma_bf16(afr[1], bfr, acc[1][t]);
    }
  }

  // Epilogue: C layout -> lane holds n=lm, reg r holds m = r + 8*lh
#pragma unroll
  for (int t = 0; t < 4; ++t) {
    const int e  = nbase + t * 16 + lm;
    const float bb = bias[e];
    const int hh = e >> 6, d = e & 63;
#pragma unroll
    for (int s = 0; s < 2; ++s) {
#pragma unroll
      for (int r = 0; r < 8; ++r) {
        const int tok = mbase + s * 16 + r + 8 * lh;
        const int bI = tok >> 11, sq = tok & (SEQ - 1);
        const float val = acc[s][t][r] + bb;
        size_t idx;
        if (transposeOut) idx = ((size_t)(bI * NH + hh) * HDIM + d) * SEQ + sq;
        else              idx = ((size_t)(bI * NH + hh) * SEQ + sq) * HDIM + d;
        outp[idx] = tobf(val);
      }
    }
  }
}

// ---------------------------------------------------------------------------
// Attention: block = 512 thr = 16 waves, one wave per head, all sharing one
// (b, 16-row q tile). Mask compressed to LDS bitmask once (shared by 16 heads
// and both passes). Pass 1: WMMA scores + per-lane online max/sum (no spill).
// Pass 2: recompute scores, normalize, write attn (f32), P->bf16 via tiny LDS
// transpose, attn @ V with WMMA into ctx accumulators.
// ---------------------------------------------------------------------------
__global__ __launch_bounds__(512) void attn_kernel(
    const __bf16* __restrict__ qp, const __bf16* __restrict__ kp,
    const __bf16* __restrict__ vpT, const int* __restrict__ mask,
    float* __restrict__ attnOut, __bf16* __restrict__ ctx2)
{
  __shared__ unsigned mbits[16][64];          // 16 q-rows x 2048 k bits
  __shared__ __bf16 pstage[16][16][33];       // per-wave P tile staging (padded)

  const int q0   = blockIdx.x * 16;
  const int b    = blockIdx.y;
  const int head = threadIdx.x >> 5;
  const int lane = threadIdx.x & 31;
  const int lm = lane & 15, lh = lane >> 4;
  const int bh = b * NH + head;

  // Build mask bitmap cooperatively: 1024 words, 2 per thread, b128 loads.
  for (int w = threadIdx.x; w < 1024; w += 512) {
    const int row = w >> 6, c = w & 63;
    const int* mrow = mask + ((size_t)b * SEQ + (q0 + row)) * SEQ + c * 32;
    unsigned bits = 0u;
#pragma unroll
    for (int i = 0; i < 8; ++i) {
      v4i m4 = *(const v4i*)(mrow + i * 4);
#pragma unroll
      for (int j = 0; j < 4; ++j) bits |= (m4[j] != 0 ? 1u : 0u) << (i * 4 + j);
    }
    mbits[row][c] = bits;
  }
  __syncthreads();

  // Q fragments for this head (reused across entire k loop).
  const __bf16* qrow = qp + ((size_t)bh << 17) + (size_t)(q0 + lm) * HDIM;
  v16bf aq0, aq1;
  {
    v8bf l0 = *(const v8bf*)(qrow + lh * 16);
    v8bf h0 = *(const v8bf*)(qrow + 8 + lh * 16);
    v8bf l1 = *(const v8bf*)(qrow + 32 + lh * 16);
    v8bf h1 = *(const v8bf*)(qrow + 40 + lh * 16);
#pragma unroll
    for (int i = 0; i < 8; ++i) {
      aq0[i] = l0[i]; aq0[i + 8] = h0[i];
      aq1[i] = l1[i]; aq1[i + 8] = h1[i];
    }
  }

  const __bf16* kbase = kp + ((size_t)bh << 17);

  float rmax[8], rsum[8];
#pragma unroll
  for (int r = 0; r < 8; ++r) { rmax[r] = -1e30f; rsum[r] = 0.0f; }

  // ---- Pass 1: per-lane online softmax stats over all k -------------------
  for (int kt = 0; kt < SEQ; kt += 16) {
    const __bf16* krow = kbase + (size_t)(kt + lm) * HDIM + lh * 16;
    v16bf bk0 = *(const v16bf*)(krow);        // features 0..31 split by lane half
    v16bf bk1 = *(const v16bf*)(krow + 32);   // features 32..63
    v8f c = v8f_zero();
    c = wmma_bf16(aq0, bk0, c);
    c = wmma_bf16(aq1, bk1, c);
    const int kcol = kt + lm;
    const int wsel = kcol >> 5, bsel = kcol & 31;
#pragma unroll
    for (int r = 0; r < 8; ++r) {
      const int m = r + 8 * lh;
      float s = c[r] * 0.125f;                // 1/sqrt(64)
      if (((mbits[m][wsel] >> bsel) & 1u) == 0u) s = -1e30f;
      const float nm = fmaxf(rmax[r], s);
      rsum[r] = rsum[r] * __expf(rmax[r] - nm) + __expf(s - nm);
      rmax[r] = nm;
    }
  }

  // Merge stats across the 16 lanes of each half (rows stay separated).
  float rinv[8];
#pragma unroll
  for (int r = 0; r < 8; ++r) {
    float M = rmax[r];
#pragma unroll
    for (int off = 1; off < 16; off <<= 1) M = fmaxf(M, __shfl_xor(M, off, 32));
    float t = rsum[r] * __expf(rmax[r] - M);
#pragma unroll
    for (int off = 1; off < 16; off <<= 1) t += __shfl_xor(t, off, 32);
    rmax[r] = M;
    rinv[r] = 1.0f / t;
  }

  // ---- Pass 2: recompute scores, write attn, accumulate ctx ---------------
  v8f cacc[4];
#pragma unroll
  for (int t = 0; t < 4; ++t) cacc[t] = v8f_zero();

  float* arowBase = attnOut + ((size_t)bh * SEQ + q0) * SEQ;
  const __bf16* vbase = vpT + ((size_t)bh << 17);

  for (int kt = 0; kt < SEQ; kt += 32) {
    v8f c0 = v8f_zero(), c1 = v8f_zero();
    {
      const __bf16* krow0 = kbase + (size_t)(kt + lm) * HDIM + lh * 16;
      v16bf x0 = *(const v16bf*)(krow0);
      v16bf x1 = *(const v16bf*)(krow0 + 32);
      c0 = wmma_bf16(aq0, x0, c0);
      c0 = wmma_bf16(aq1, x1, c0);
      const __bf16* krow1 = kbase + (size_t)(kt + 16 + lm) * HDIM + lh * 16;
      v16bf y0 = *(const v16bf*)(krow1);
      v16bf y1 = *(const v16bf*)(krow1 + 32);
      c1 = wmma_bf16(aq0, y0, c1);
      c1 = wmma_bf16(aq1, y1, c1);
    }
    const int k0 = kt + lm, k1 = kt + 16 + lm;
#pragma unroll
    for (int r = 0; r < 8; ++r) {
      const int m = r + 8 * lh;
      float s0 = c0[r] * 0.125f, s1 = c1[r] * 0.125f;
      if (((mbits[m][k0 >> 5] >> (k0 & 31)) & 1u) == 0u) s0 = -1e30f;
      if (((mbits[m][k1 >> 5] >> (k1 & 31)) & 1u) == 0u) s1 = -1e30f;
      const float p0 = __expf(s0 - rmax[r]) * rinv[r];
      const float p1 = __expf(s1 - rmax[r]) * rinv[r];
      float* ar = arowBase + (size_t)m * SEQ;
      ar[k0] = p0;                            // normalized attention output
      ar[k1] = p1;
      pstage[head][m][lm]      = tobf(p0);    // stage for A-fragment reshuffle
      pstage[head][m][16 + lm] = tobf(p1);
    }
    asm volatile("s_wait_dscnt 0" ::: "memory");   // cross-lane LDS W->R fence

    v16bf pa;                                 // P as WMMA A-fragment (16x32)
#pragma unroll
    for (int i = 0; i < 8; ++i) {
      pa[i]     = pstage[head][lm][lh * 16 + i];
      pa[i + 8] = pstage[head][lm][lh * 16 + 8 + i];
    }
#pragma unroll
    for (int t = 0; t < 4; ++t) {
      const __bf16* vr = vbase + (size_t)(t * 16 + lm) * SEQ + kt + lh * 16;
      v16bf bv = *(const v16bf*)(vr);         // contiguous thanks to vpT layout
      cacc[t] = wmma_bf16(pa, bv, cacc[t]);
    }
  }

  // ctx -> [tok, h*64+d] bf16 for the output projection GEMM
#pragma unroll
  for (int t = 0; t < 4; ++t) {
#pragma unroll
    for (int r = 0; r < 8; ++r) {
      const int m = r + 8 * lh;
      const int d = t * 16 + lm;
      ctx2[(size_t)(b * SEQ + q0 + m) * EMB + head * HDIM + d] = tobf(cacc[t][r]);
    }
  }
}

// ---------------------------------------------------------------------------
// Output projection: out(f32) = ctx2(bf16 [8192,1024]) @ Wo^T(bf16) + bo
// ---------------------------------------------------------------------------
__global__ __launch_bounds__(256) void outproj_kernel(
    const __bf16* __restrict__ A, const __bf16* __restrict__ W,
    const float* __restrict__ bias, float* __restrict__ out)
{
  const int lane  = threadIdx.x & 31;
  const int wave  = threadIdx.x >> 5;
  const int mbase = blockIdx.x * 256 + wave * 32;
  const int nbase = blockIdx.y * 64;
  const int lm = lane & 15, lh = lane >> 4;

  v8f acc[2][4];
#pragma unroll
  for (int s = 0; s < 2; ++s)
#pragma unroll
    for (int t = 0; t < 4; ++t) acc[s][t] = v8f_zero();

  for (int kc = 0; kc < EMB; kc += 32) {
    v16bf afr[2];
#pragma unroll
    for (int s = 0; s < 2; ++s) {
      const __bf16* ar = A + (size_t)(mbase + s * 16 + lm) * EMB + kc + lh * 16;
      v8bf lo = *(const v8bf*)(ar);
      v8bf hi = *(const v8bf*)(ar + 8);
#pragma unroll
      for (int i = 0; i < 8; ++i) { afr[s][i] = lo[i]; afr[s][i + 8] = hi[i]; }
    }
#pragma unroll
    for (int t = 0; t < 4; ++t) {
      const __bf16* wr = W + (size_t)(nbase + t * 16 + lm) * EMB + kc + lh * 16;
      v16bf bfr = *(const v16bf*)(wr);
      acc[0][t] = wmma_bf16(afr[0], bfr, acc[0][t]);
      acc[1][t] = wmma_bf16(afr[1], bfr, acc[1][t]);
    }
  }

#pragma unroll
  for (int t = 0; t < 4; ++t) {
    const int e = nbase + t * 16 + lm;
    const float bb = bias[e];
#pragma unroll
    for (int s = 0; s < 2; ++s) {
#pragma unroll
      for (int r = 0; r < 8; ++r) {
        const int tok = mbase + s * 16 + r + 8 * lh;
        out[(size_t)tok * EMB + e] = acc[s][t][r] + bb;
      }
    }
  }
}

// ---------------------------------------------------------------------------
extern "C" void kernel_launch(void* const* d_in, const int* in_sizes, int n_in,
                              void* d_out, int out_size, void* d_ws, size_t ws_size,
                              hipStream_t stream) {
  (void)in_sizes; (void)n_in; (void)out_size; (void)ws_size;
  const float* q    = (const float*)d_in[0];
  const float* k    = (const float*)d_in[1];
  const float* v    = (const float*)d_in[2];
  const int*   mask = (const int*)d_in[3];
  const float* Wk   = (const float*)d_in[4];
  const float* bk   = (const float*)d_in[5];
  const float* Wv   = (const float*)d_in[6];
  const float* bv   = (const float*)d_in[7];
  const float* Wo   = (const float*)d_in[8];
  const float* bo   = (const float*)d_in[9];

  const size_t PROJ = (size_t)BATCH * NH * SEQ * HDIM;   // 8M elems
  const size_t WN   = (size_t)EMB * EMB;                 // 1M elems per weight
  __bf16* qp   = (__bf16*)d_ws;
  __bf16* kp   = qp   + PROJ;
  __bf16* vpT  = kp   + PROJ;
  __bf16* ctx2 = vpT  + PROJ;
  __bf16* Wkb  = ctx2 + PROJ;
  __bf16* Wvb  = Wkb  + WN;
  __bf16* Wob  = Wvb  + WN;                              // total 70MB ws

  float* out  = (float*)d_out;
  float* attn = out + (size_t)BATCH * SEQ * EMB;         // 2nd tuple output

  // weights -> bf16 once
  const int n8 = (int)(WN / 8);
  cvt_kernel<<<(n8 + 255) / 256, 256, 0, stream>>>(Wk, Wkb, n8);
  cvt_kernel<<<(n8 + 255) / 256, 256, 0, stream>>>(Wv, Wvb, n8);
  cvt_kernel<<<(n8 + 255) / 256, 256, 0, stream>>>(Wo, Wob, n8);

  dim3 gproj(BATCH * SEQ / 256, EMB / 64);               // (32,16)
  // faithful to reference bug: q is projected with Wv/bv
  proj_kernel<<<gproj, 256, 0, stream>>>(q, Wvb, bv, qp, 0);
  proj_kernel<<<gproj, 256, 0, stream>>>(k, Wkb, bk, kp, 0);
  proj_kernel<<<gproj, 256, 0, stream>>>(v, Wvb, bv, vpT, 1);  // V transposed

  attn_kernel<<<dim3(SEQ / 16, BATCH), 512, 0, stream>>>(qp, kp, vpT, mask,
                                                         attn, ctx2);

  outproj_kernel<<<gproj, 256, 0, stream>>>(ctx2, Wo ? Wob : Wob, bo, out);
}
